// ScaledDotProductAttention_70119636074661
// MI455X (gfx1250) — matrix operand order repaired and checked by reading, not compiled, not run on hardware
//
#include <hip/hip_runtime.h>

// ---- CDNA5 (gfx1250) wave32 WMMA types ----
typedef __attribute__((ext_vector_type(16))) __bf16 v16bf;
typedef __attribute__((ext_vector_type(8)))  float  v8f;

#define B_   2
#define H_   16
#define S_   2048
#define DK_  64
#define WAVES_PER_BLOCK 4
#define KV_STEP 32                       // keys per staged tile
#define N_TILES (S_ / KV_STEP)           // 64
#define TILE_F  (KV_STEP * DK_)          // 2048 floats = 8KB per K or V tile

#define LOG2E   1.4426950408889634f
#define SSCALE  (0.125f * LOG2E)         // (1/sqrt(64)) * log2(e): softmax in exp2 domain
#define NEG2    (-1000000000.0f * LOG2E) // masked logit in exp2 domain

// ---------------- TDM (Tensor Data Mover) support --------------------------
#ifndef USE_TDM
#if defined(__has_builtin)
#if __has_builtin(__builtin_amdgcn_tensor_load_to_lds) && __has_builtin(__builtin_amdgcn_s_wait_tensorcnt)
#define USE_TDM 1
#else
#define USE_TDM 0
#endif
#else
#define USE_TDM 0
#endif
#endif

#if USE_TDM
typedef unsigned int tdm_u32x4 __attribute__((ext_vector_type(4)));
typedef int          tdm_i32x4 __attribute__((ext_vector_type(4)));
typedef int          tdm_i32x8 __attribute__((ext_vector_type(8)));

// 1-D contiguous global->LDS DMA: nElems f32 elements starting at gptr.
// D# packing per CDNA5 ISA (group0 + group1).
__device__ __forceinline__ void tdm_load_1d(unsigned int lds_byte_addr,
                                            const void* gptr,
                                            unsigned int nElems) {
  const unsigned long long ga = (unsigned long long)(size_t)gptr;
  tdm_u32x4 g0;
  g0[0] = 1u;                                           // count=1 (valid user D#)
  g0[1] = lds_byte_addr;                                // lds_addr
  g0[2] = (unsigned int)(ga & 0xFFFFFFFFu);             // global_addr[31:0]
  g0[3] = (unsigned int)((ga >> 32) & 0x01FFFFFFu)      // global_addr[56:32]
        | (2u << 30);                                   // type=2 ("image")
  tdm_i32x8 g1;
  g1[0] = (int)(2u << 16);                              // wg_mask=0, data_size=2 (4B)
  g1[1] = (int)((nElems & 0xFFFFu) << 16);              // tensor_dim0[15:0]
  g1[2] = (int)((nElems >> 16) & 0xFFFFu);              // tensor_dim0[31:16], tensor_dim1=0
  g1[3] = (int)((nElems & 0xFFFFu) << 16);              // tile_dim0 = nElems (<=65535)
  g1[4] = 0;                                            // tile_dim1=0, tile_dim2=0 (1-D)
  g1[5] = (int)nElems;                                  // tensor_dim0_stride (unused for 1-D)
  g1[6] = 0;
  g1[7] = 0;
  tdm_i32x4 z4 = {0, 0, 0, 0};
#if defined(__clang_major__) && (__clang_major__ >= 23)
  tdm_i32x8 z8 = {0, 0, 0, 0, 0, 0, 0, 0};
  __builtin_amdgcn_tensor_load_to_lds(g0, g1, z4, z4, z8, 0);
#else
  __builtin_amdgcn_tensor_load_to_lds(g0, g1, z4, z4, 0);
#endif
}
#endif  // USE_TDM

// Fallback / portable cooperative tile copy (block-wide float4 loads)
__device__ __forceinline__ void coop_copy(float* dst, const float* src,
                                          int nFloats, int tid, int nthr) {
  const float4* s = (const float4*)src;
  float4*       d = (float4*)dst;
#pragma unroll 4
  for (int i = tid; i < nFloats / 4; i += nthr) d[i] = s[i];
}

// ---------------- fragment packing helpers ---------------------------------
__device__ __forceinline__ v16bf pack16(float4 a, float4 b, float4 c, float4 d) {
  v16bf r;
  r[0]=(__bf16)a.x;  r[1]=(__bf16)a.y;  r[2]=(__bf16)a.z;  r[3]=(__bf16)a.w;
  r[4]=(__bf16)b.x;  r[5]=(__bf16)b.y;  r[6]=(__bf16)b.z;  r[7]=(__bf16)b.w;
  r[8]=(__bf16)c.x;  r[9]=(__bf16)c.y;  r[10]=(__bf16)c.z; r[11]=(__bf16)c.w;
  r[12]=(__bf16)d.x; r[13]=(__bf16)d.y; r[14]=(__bf16)d.z; r[15]=(__bf16)d.w;
  return r;
}

// A fragment (16x32 bf16), row-major f32 source, rstride floats per row.
// ISA 16-bit A layout: lane M=lane&15; lo half K=2j(+hi*8) pairs, j>=4 at +16.
// Each lane's data = two contiguous 8-float runs -> 4x float4 loads.
__device__ __forceinline__ v16bf load_A_f32(const float* src, int rstride, int lane) {
  const int m  = lane & 15;
  const int hi = lane >> 4;
  const float* p = src + m * rstride + hi * 8;
  const float4* q0 = (const float4*)p;          // K = hi*8 + 0..7
  const float4* q1 = (const float4*)(p + 16);   // K = 16 + hi*8 + 0..7
  return pack16(q0[0], q0[1], q1[0], q1[1]);
}

// B fragment (32x16 bf16), table order: K = hi*8 + 0..7 and 16 + hi*8 + 0..7:
// caller passes src16 already offset to (row n, kk = hi*8) -> 4x float4 loads.
__device__ __forceinline__ v16bf load_B_contig(const float* src16) {
  const float4* p = (const float4*)src16;
  return pack16(p[0], p[1], p[4], p[5]);
}

// B fragment for V from LDS: element (kk,n) = vb[kk*DK + dc*16 + n], strided.
__device__ __forceinline__ v16bf load_Bv_lds(const float* vb, int dc, int lane) {
  const int n  = lane & 15;
  const int hi = lane >> 4;
  const float* p = vb + dc * 16 + n;
  v16bf bm;
#pragma unroll
  for (int j = 0; j < 8; ++j) {
    const int kk = 2 * j + (j >= 4 ? 8 : 0) + hi * 8;
    bm[2 * j]     = (__bf16)p[kk * DK_];
    bm[2 * j + 1] = (__bf16)p[(kk + 1) * DK_];
  }
  return bm;
}

__global__ __launch_bounds__(WAVES_PER_BLOCK * 32)
void sdpa_fused_bf16wmma_kernel(const float* __restrict__ q,
                                const float* __restrict__ k,
                                const float* __restrict__ v,
                                const int*   __restrict__ mask,
                                float* __restrict__ out_ctx,
                                float* __restrict__ out_attn) {
  // Shared K/V staging (block-wide, double buffered) + per-wave P staging.
  __shared__ __align__(16) float kbuf[2][TILE_F];                 // 16 KB
  __shared__ __align__(16) float vbuf[2][TILE_F];                 // 16 KB
  __shared__ __align__(16) float pbuf[WAVES_PER_BLOCK][16][32];   // 8 KB

  const int tid  = threadIdx.x;
  const int lane = tid & 31;
  const int wave = tid >> 5;
  const int tile = blockIdx.x * WAVES_PER_BLOCK + wave;   // 0..4095
  const int bh   = tile >> 7;                             // b*H + h (same for whole block)
  const int qt   = tile & 127;
  const int b    = bh >> 4;
  const int qbase = qt * 16;

  const float* qp = q + ((size_t)bh * S_ + qbase) * DK_;
  const float* kp = k + (size_t)bh * S_ * DK_;
  const float* vp = v + (size_t)bh * S_ * DK_;
  const int*   mp = mask + ((size_t)b * S_ + qbase) * S_;
  float* attn_p   = out_attn + ((size_t)bh * S_ + qbase) * S_;

  const int hi = lane >> 4;
  const int ln = lane & 15;

#if USE_TDM
  const unsigned int koff[2] = {(unsigned int)(size_t)&kbuf[0][0],
                                (unsigned int)(size_t)&kbuf[1][0]};
  const unsigned int voff[2] = {(unsigned int)(size_t)&vbuf[0][0],
                                (unsigned int)(size_t)&vbuf[1][0]};
#endif

  // Q fragments (dk chunks 0..31 / 32..63): loaded once, reused 2*128 times.
  const v16bf aq0 = load_A_f32(qp,      DK_, lane);
  const v16bf aq1 = load_A_f32(qp + 32, DK_, lane);

  // ============== Pass 1: per-lane online softmax stats =====================
  float mrow[8], lrow[8];
#pragma unroll
  for (int r = 0; r < 8; ++r) { mrow[r] = -INFINITY; lrow[r] = 0.0f; }

#if USE_TDM
  if (wave == 0) tdm_load_1d(koff[0], kp, TILE_F);
#else
  coop_copy(kbuf[0], kp, TILE_F, tid, WAVES_PER_BLOCK * 32);
#endif

  for (int it = 0; it < N_TILES; ++it) {
    const int cur = it & 1;
#if USE_TDM
    if (wave == 0) {
      if (it + 1 < N_TILES) {
        tdm_load_1d(koff[cur ^ 1], kp + (size_t)(it + 1) * TILE_F, TILE_F);
        __builtin_amdgcn_s_wait_tensorcnt(1);   // tile `it` landed (in-order)
      } else {
        __builtin_amdgcn_s_wait_tensorcnt(0);
      }
    }
#endif
    __syncthreads();  // staged tile visible to all waves

    const float* kb = kbuf[cur];
#pragma unroll
    for (int t = 0; t < 2; ++t) {
      // B[kk][n] = K[t*16+n][c*32+kk]; per-lane two 8-float runs (table order).
      const float* krow = kb + (t * 16 + ln) * DK_ + hi * 8;
      const v16bf b0 = load_B_contig(krow);
      const v16bf b1 = load_B_contig(krow + 32);
      v8f s = {};
      s = __builtin_amdgcn_wmma_f32_16x16x32_bf16(false, aq0, false, b0, (short)0, s, false, false);
      s = __builtin_amdgcn_wmma_f32_16x16x32_bf16(false, aq1, false, b1, (short)0, s, false, false);
      const int kcol = it * KV_STEP + t * 16 + ln;
#pragma unroll
      for (int r = 0; r < 8; ++r) {
        const int qrow = r + hi * 8;
        float sv = s[r] * SSCALE;
        if (mp[(size_t)qrow * S_ + kcol] == 0) sv = NEG2;
        const float mn = fmaxf(mrow[r], sv);               // per-lane online update
        lrow[r] = lrow[r] * exp2f(mrow[r] - mn) + exp2f(sv - mn);
        mrow[r] = mn;
      }
    }
    __syncthreads();  // all reads of kbuf[cur] done -> safe to overwrite
  }

  // one-shot 16-lane merge of per-lane (m,l): rows live across lanes 0-15/16-31
#pragma unroll
  for (int r = 0; r < 8; ++r) {
    float m = mrow[r], l = lrow[r];
#pragma unroll
    for (int off = 1; off < 16; off <<= 1) {
      const float mo = __shfl_xor(m, off, 16);
      const float lo = __shfl_xor(l, off, 16);
      const float mn = fmaxf(m, mo);
      l = l * exp2f(m - mn) + lo * exp2f(mo - mn);
      m = mn;
    }
    mrow[r] = m;
    lrow[r] = 1.0f / l;  // inverse denominator
  }

  // ===== Pass 2: recompute S, emit normalized attention, accumulate P*V =====
  v8f ctx[4] = {v8f{}, v8f{}, v8f{}, v8f{}};
  float (*pl)[32] = pbuf[wave];

#if USE_TDM
  if (wave == 0) {
    tdm_load_1d(koff[0], kp, TILE_F);
    tdm_load_1d(voff[0], vp, TILE_F);
  }
#else
  coop_copy(kbuf[0], kp, TILE_F, tid, WAVES_PER_BLOCK * 32);
  coop_copy(vbuf[0], vp, TILE_F, tid, WAVES_PER_BLOCK * 32);
#endif

  for (int it = 0; it < N_TILES; ++it) {
    const int cur = it & 1;
#if USE_TDM
    if (wave == 0) {
      if (it + 1 < N_TILES) {
        tdm_load_1d(koff[cur ^ 1], kp + (size_t)(it + 1) * TILE_F, TILE_F);
        tdm_load_1d(voff[cur ^ 1], vp + (size_t)(it + 1) * TILE_F, TILE_F);
        __builtin_amdgcn_s_wait_tensorcnt(2);   // K/V of tile `it` landed
      } else {
        __builtin_amdgcn_s_wait_tensorcnt(0);
      }
    }
#endif
    __syncthreads();

    const float* kb = kbuf[cur];
    const float* vb = vbuf[cur];
#pragma unroll
    for (int t = 0; t < 2; ++t) {
      const float* krow = kb + (t * 16 + ln) * DK_ + hi * 8;
      const v16bf b0 = load_B_contig(krow);
      const v16bf b1 = load_B_contig(krow + 32);
      v8f s = {};
      s = __builtin_amdgcn_wmma_f32_16x16x32_bf16(false, aq0, false, b0, (short)0, s, false, false);
      s = __builtin_amdgcn_wmma_f32_16x16x32_bf16(false, aq1, false, b1, (short)0, s, false, false);
      const int kcol = it * KV_STEP + t * 16 + ln;
#pragma unroll
      for (int r = 0; r < 8; ++r) {
        const int qrow = r + hi * 8;
        float sv = s[r] * SSCALE;
        if (mp[(size_t)qrow * S_ + kcol] == 0) sv = NEG2;
        const float p = exp2f(sv - mrow[r]) * lrow[r];
        // streaming 536MB output: non-temporal so Q/K/V stay L2-resident
        __builtin_nontemporal_store(p, &attn_p[(size_t)qrow * S_ + kcol]);
        pl[qrow][t * 16 + ln] = p;   // stage for A-layout re-read (same-wave LDS)
      }
    }
    // P (16x32) as A fragment; V chunks as B: ctx += P * V
    const v16bf ap = load_A_f32(&pl[0][0], 32, lane);
#pragma unroll
    for (int dc = 0; dc < 4; ++dc) {
      const v16bf bv = load_Bv_lds(vb, dc, lane);
      ctx[dc] = __builtin_amdgcn_wmma_f32_16x16x32_bf16(false, ap, false, bv, (short)0, ctx[dc], false, false);
    }
    __syncthreads();  // all reads of kbuf/vbuf[cur] done -> safe to overwrite
  }

  // context store: element (m = r + hi*8, d = dc*16 + ln)
#pragma unroll
  for (int r = 0; r < 8; ++r) {
    const int qrow = r + hi * 8;
#pragma unroll
    for (int dc = 0; dc < 4; ++dc) {
      out_ctx[((size_t)bh * S_ + qbase + qrow) * DK_ + dc * 16 + ln] = ctx[dc][r];
    }
  }
}

extern "C" void kernel_launch(void* const* d_in, const int* in_sizes, int n_in,
                              void* d_out, int out_size, void* d_ws, size_t ws_size,
                              hipStream_t stream) {
  const float* q    = (const float*)d_in[0];
  const float* k    = (const float*)d_in[1];
  const float* v    = (const float*)d_in[2];
  const int*   mask = (const int*)d_in[3];

  float* ctx  = (float*)d_out;                     // [B,H,S,DK]
  float* attn = ctx + (size_t)B_ * H_ * S_ * DK_;  // [B,H,S,S]

  const int n_tiles = B_ * H_ * (S_ / 16);         // 4096 wave tiles
  dim3 grid(n_tiles / WAVES_PER_BLOCK);            // 1024 blocks (each block: one bh slice)
  dim3 block(WAVES_PER_BLOCK * 32);                // 4 wave32s
  sdpa_fused_bf16wmma_kernel<<<grid, block, 0, stream>>>(q, k, v, mask, ctx, attn);
}
